// GNN_2_395136991891
// MI455X (gfx1250) — compile-verified, matrix operand order checked by reading, not run on hardware
//
#include <hip/hip_runtime.h>
#include <hip/hip_bf16.h>

typedef __attribute__((ext_vector_type(16))) _Float16 v16h;
typedef __attribute__((ext_vector_type(8)))  _Float16 v8h;
typedef __attribute__((ext_vector_type(8)))  float    v8f;

#define FEATS 64
#define LDS_PITCH 72   // 64 + 8 halfs pad; 144B row stride keeps 16B alignment

// ---------------------------------------------------------------------------
// K1: deg[n] = 1.0f  (self-loop weight)
// ---------------------------------------------------------------------------
__global__ void gcn_deg_init(float* __restrict__ deg, int n_nodes) {
  int i = blockIdx.x * blockDim.x + threadIdx.x;
  if (i < n_nodes) deg[i] = 1.0f;
}

// ---------------------------------------------------------------------------
// K2: deg[col[e]] += ew[e]   (weighted in-degree scatter)
// ---------------------------------------------------------------------------
__global__ void gcn_deg_scatter(const long long* __restrict__ col,
                                const float* __restrict__ ew,
                                float* __restrict__ deg, int n_edges) {
  int e = blockIdx.x * blockDim.x + threadIdx.x;
  if (e < n_edges) {
    atomicAdd(&deg[(int)col[e]], ew[e]);
  }
}

// ---------------------------------------------------------------------------
// Pack four float4s into a v16h fragment (lowered to v_cvt_pk_f16_f32 pairs).
// ---------------------------------------------------------------------------
__device__ __forceinline__ v16h pack16(float4 a, float4 b, float4 c, float4 d) {
  v16h r;
  r[0]  = (_Float16)a.x; r[1]  = (_Float16)a.y; r[2]  = (_Float16)a.z; r[3]  = (_Float16)a.w;
  r[4]  = (_Float16)b.x; r[5]  = (_Float16)b.y; r[6]  = (_Float16)b.z; r[7]  = (_Float16)b.w;
  r[8]  = (_Float16)c.x; r[9]  = (_Float16)c.y; r[10] = (_Float16)c.z; r[11] = (_Float16)c.w;
  r[12] = (_Float16)d.x; r[13] = (_Float16)d.y; r[14] = (_Float16)d.z; r[15] = (_Float16)d.w;
  return r;
}

// ---------------------------------------------------------------------------
// K3 (WMMA): h0 = (x @ W)[:, 0] via v_wmma_f32_16x16x32_f16.
// One wave per 16-row tile of x; two K=32 WMMAs cover K=64.
// A (16-bit 16x32 layout): lane m = row; half slot i -> K = i + (i<8?0:8) + 8*g,
// i.e. two contiguous 8-float runs per lane per fragment -> global_load_b128 x2.
// B (32x16): column-striped; staged once per block into LDS transposed
// (Wt[col][k]) so each lane's fragment is four contiguous ds_load_b128 runs.
// Only column N=0 of D (lanes 0 and 16) is stored.
// ---------------------------------------------------------------------------
__global__ void gcn_h0_wmma(const float* __restrict__ x,
                            const float* __restrict__ W,
                            float* __restrict__ h0, int n_nodes) {
  __shared__ _Float16 Wt[16][LDS_PITCH];   // W[:,0:16] transposed, f16 (2.25 KB)

  // Cooperative stage: 1024 elements over 256 threads.
  for (int e = threadIdx.x; e < 16 * 64; e += blockDim.x) {
    const int c = e >> 6;        // output column 0..15
    const int k = e & 63;        // K index 0..63
    Wt[c][k] = (_Float16)W[k * FEATS + c];
  }
  __syncthreads();

  const int wave = threadIdx.x >> 5;
  const int lane = threadIdx.x & 31;
  const int tile = blockIdx.x * (blockDim.x >> 5) + wave;
  const int R    = tile * 16;
  if (R >= n_nodes) return;                // wave-uniform exit (after barrier)

  const int g = lane >> 4;                 // lane half: K sub-block select
  const int m = lane & 15;                 // row (A) / column (B)

  int row = R + m;
  if (row >= n_nodes) row = R;             // branchless clamp, EXEC stays full
  const float4* xr4 = (const float4*)(x + (size_t)row * FEATS);

  // A fragments: k runs [g*8, g*8+8) and [16+g*8, 16+g*8+8) (+32 for a1).
  const float4 q0 = xr4[2 * g];
  const float4 q1 = xr4[2 * g + 1];
  const float4 q2 = xr4[4 + 2 * g];
  const float4 q3 = xr4[4 + 2 * g + 1];
  const float4 q4 = xr4[8 + 2 * g];
  const float4 q5 = xr4[8 + 2 * g + 1];
  const float4 q6 = xr4[12 + 2 * g];
  const float4 q7 = xr4[12 + 2 * g + 1];
  const v16h a0 = pack16(q0, q1, q2, q3);
  const v16h a1 = pack16(q4, q5, q6, q7);

  // B fragments from LDS: column m, same K mapping -> 4x ds_load_b128.
  const _Float16* wrow = &Wt[m][0];
  const v8h w0 = *(const v8h*)(wrow + g * 8);
  const v8h w1 = *(const v8h*)(wrow + 16 + g * 8);
  const v8h w2 = *(const v8h*)(wrow + 32 + g * 8);
  const v8h w3 = *(const v8h*)(wrow + 48 + g * 8);
  const v16h b0 = __builtin_shufflevector(w0, w1, 0,1,2,3,4,5,6,7,8,9,10,11,12,13,14,15);
  const v16h b1 = __builtin_shufflevector(w2, w3, 0,1,2,3,4,5,6,7,8,9,10,11,12,13,14,15);

  v8f c = {};
  c = __builtin_amdgcn_wmma_f32_16x16x32_f16(false, a0, false, b0,
                                             (short)0, c, false, false);
  c = __builtin_amdgcn_wmma_f32_16x16x32_f16(false, a1, false, b1,
                                             (short)0, c, false, false);

  // D layout: VGPR r, lanes 0-15 -> (M=r, N=lane); lanes 16-31 -> (M=r+8, N=lane-16).
  // Column N=0: lane 0 holds rows R..R+7 in c[0..7], lane 16 holds R+8..R+15.
  if (m == 0) {
    float* dst = h0 + R + g * 8;
    if (R + 16 <= n_nodes) {               // wave-uniform full-tile fast path
      *(float4*)(dst)     = make_float4(c[0], c[1], c[2], c[3]);
      *(float4*)(dst + 4) = make_float4(c[4], c[5], c[6], c[7]);
    } else {
      for (int r = 0; r < 8; ++r)
        if (R + g * 8 + r < n_nodes) dst[r] = c[r];
    }
  }
}

// ---------------------------------------------------------------------------
// K4: dis = rsqrt(deg) (in place over deg); out0 = dis^2 * h0 (self-loop term)
// ---------------------------------------------------------------------------
__global__ void gcn_norm_selfloop(float* __restrict__ degdis,
                                  const float* __restrict__ h0,
                                  float* __restrict__ out0, int n_nodes) {
  int i = blockIdx.x * blockDim.x + threadIdx.x;
  if (i < n_nodes) {
    const float d  = degdis[i];
    const float di = (d > 0.0f) ? rsqrtf(d) : 0.0f;
    degdis[i] = di;
    out0[i]   = di * di * h0[i];
  }
}

// ---------------------------------------------------------------------------
// K5: edge scatter  out0[col] += dis[row]*ew*dis[col]*h0[row]
// Edge arrays (~64 MB) largely L2-resident after K2; dis/h0/out0 (400 KB each)
// stay in L2, so gathers and global_atomic_add_f32 are L2-speed.
// ---------------------------------------------------------------------------
__global__ void gcn_edge_scatter(const long long* __restrict__ row,
                                 const long long* __restrict__ col,
                                 const float* __restrict__ ew,
                                 const float* __restrict__ dis,
                                 const float* __restrict__ h0,
                                 float* __restrict__ out0, int n_edges) {
  int e = blockIdx.x * blockDim.x + threadIdx.x;
  if (e < n_edges) {
    const int r = (int)row[e];
    const int c = (int)col[e];
    const float v = dis[r] * ew[e] * dis[c] * h0[r];
    atomicAdd(&out0[c], v);
  }
}

// ---------------------------------------------------------------------------
// K6: o[n] = out0[n] + b[0] - x[n,0]   (in place on d_out)
// ---------------------------------------------------------------------------
__global__ void gcn_finalize(float* __restrict__ out0,
                             const float* __restrict__ x,
                             const float* __restrict__ b, int n_nodes) {
  int i = blockIdx.x * blockDim.x + threadIdx.x;
  if (i < n_nodes) {
    out0[i] = out0[i] + b[0] - x[(size_t)i * FEATS];
  }
}

// ---------------------------------------------------------------------------
extern "C" void kernel_launch(void* const* d_in, const int* in_sizes, int n_in,
                              void* d_out, int out_size, void* d_ws, size_t ws_size,
                              hipStream_t stream) {
  const float*     x  = (const float*)d_in[0];          // [N, 64] f32
  const long long* ei = (const long long*)d_in[1];      // [2, E] int64
  const float*     ew = (const float*)d_in[2];          // [E] f32
  const float*     W  = (const float*)d_in[3];          // [64, 64] f32
  const float*     b  = (const float*)d_in[4];          // [64] f32

  const int n_nodes = in_sizes[0] / FEATS;
  const int n_edges = in_sizes[2];
  const long long* row = ei;            // edge_index[0] = source
  const long long* col = ei + n_edges;  // edge_index[1] = target

  float* degdis = (float*)d_ws;                 // [N] deg -> dis (in place)
  float* h0     = (float*)d_ws + n_nodes;       // [N]
  float* out0   = (float*)d_out;                // [N] accumulate in output

  const int T = 256;
  const int gN = (n_nodes + T - 1) / T;
  const int gE = (n_edges + T - 1) / T;
  const int tiles = (n_nodes + 15) / 16;        // one wave per 16-row tile
  const int gW = (tiles + 7) / 8;               // 8 waves per 256-thread block

  gcn_deg_init     <<<gN, T, 0, stream>>>(degdis, n_nodes);
  gcn_deg_scatter  <<<gE, T, 0, stream>>>(col, ew, degdis, n_edges);
  gcn_h0_wmma      <<<gW, T, 0, stream>>>(x, W, h0, n_nodes);
  gcn_norm_selfloop<<<gN, T, 0, stream>>>(degdis, h0, out0, n_nodes);
  gcn_edge_scatter <<<gE, T, 0, stream>>>(row, col, ew, degdis, h0, out0, n_edges);
  gcn_finalize     <<<gN, T, 0, stream>>>(out0, x, b, n_nodes);
}
